// NMPN_44693429682679
// MI455X (gfx1250) — matrix operand order repaired and checked by reading, not compiled, not run on hardware
//
#include <hip/hip_runtime.h>
#include <hip/hip_bf16.h>

#define N_ATOMS   50000
#define N_BONDS   120000
#define MAX_NB    6
#define ATOM_FDIM 39
#define BOND_FDIM 11
#define HIDDEN    512
#define DEPTH     4
#define KDIM      523           // HIDDEN + BOND_FDIM
#define KPAD      544           // 17 * 32, zero padded

typedef __attribute__((ext_vector_type(16))) _Float16 v16h;
typedef __attribute__((ext_vector_type(8)))  _Float16 v8h;
typedef __attribute__((ext_vector_type(8)))  float    v8f;

// ---------------------------------------------------------------------------
// h0 = ReLU(fatoms @ W_nin^T)   [N_ATOMS x HIDDEN], fp32
// one block per atom; fatoms row cached in LDS; W_nin (80KB) is cache-resident
// ---------------------------------------------------------------------------
__global__ void __launch_bounds__(256)
nin_kernel(const float* __restrict__ fatoms,
           const float* __restrict__ W_nin,
           float* __restrict__ h0out)
{
    __shared__ float fa[ATOM_FDIM];
    const int a = blockIdx.x;
    const int t = threadIdx.x;
    if (t < ATOM_FDIM) fa[t] = fatoms[(size_t)a * ATOM_FDIM + t];
    __syncthreads();
    for (int h = t; h < HIDDEN; h += 256) {
        const float* w = W_nin + (size_t)h * ATOM_FDIM;
        float s = 0.f;
        #pragma unroll
        for (int k = 0; k < ATOM_FDIM; ++k) s = fmaf(w[k], fa[k], s);
        h0out[(size_t)a * HIDDEN + h] = s > 0.f ? s : 0.f;
    }
}

// ---------------------------------------------------------------------------
// W_node (fp32, [HIDDEN][KDIM]) -> f16 [HIDDEN][KPAD], zero padded K tail
// ---------------------------------------------------------------------------
__global__ void __launch_bounds__(256)
wconv_kernel(const float* __restrict__ W_node, _Float16* __restrict__ Wf)
{
    int i = blockIdx.x * 256 + threadIdx.x;
    if (i >= HIDDEN * KPAD) return;
    int n = i / KPAD, k = i % KPAD;
    Wf[i] = (k < KDIM) ? (_Float16)W_node[(size_t)n * KDIM + k] : (_Float16)0.f;
}

// ---------------------------------------------------------------------------
// Gather-sum: NEI[a][k] (f16, KPAD cols) = sum over 6 neighbor bonds of
//   msg[b] = concat( b==0 ? 0 : H[in_n[b-1]] , fbonds[b] )
// One wave32 per atom; fp32 accumulation, f16 store.
// ---------------------------------------------------------------------------
__global__ void __launch_bounds__(256)
gather_kernel(const float* __restrict__ H,
              const float* __restrict__ fbonds,
              const int*   __restrict__ aout,
              const int*   __restrict__ in_n,
              _Float16*    __restrict__ NEI)
{
    const int lane = threadIdx.x & 31;
    const int wave = threadIdx.x >> 5;
    const int a = blockIdx.x * 8 + wave;
    if (a >= N_ATOMS) return;

    int b[MAX_NB], src[MAX_NB];
    #pragma unroll
    for (int j = 0; j < MAX_NB; ++j) {
        b[j] = aout[(size_t)a * MAX_NB + j];
        src[j] = (b[j] == 0) ? -1 : in_n[b[j] - 1];
    }

    _Float16* nr = NEI + (size_t)a * KPAD;

    // hidden-state columns [0, 512)
    for (int c = lane; c < HIDDEN; c += 32) {
        float s = 0.f;
        #pragma unroll
        for (int j = 0; j < MAX_NB; ++j)
            if (src[j] >= 0) s += H[(size_t)src[j] * HIDDEN + c];
        nr[c] = (_Float16)s;
    }

    // bond-feature columns [512, 523) + zero pad [523, 544)
    float s = 0.f;
    if (lane < BOND_FDIM) {
        #pragma unroll
        for (int j = 0; j < MAX_NB; ++j)
            s += fbonds[(size_t)b[j] * BOND_FDIM + lane];
    }
    nr[HIDDEN + lane] = (_Float16)s;
}

// ---------------------------------------------------------------------------
// H = ReLU(h0 + NEI @ Wf^T) via v_wmma_f32_16x16x32_f16.
// One wave per 16x16 output tile; K loop = KPAD/32 = 17 WMMAs.
// A layout (16-bit 16x32): lanes 0-15 -> K[0..15]; lanes 16-31 ->
// K[16..31]; row M = lane&15. B layout (32x16): lanes 0-15 hold
// K 0..15, lanes 16-31 hold K 16..31; col N = lane&15.
// ---------------------------------------------------------------------------
__global__ void __launch_bounds__(256)
node_gemm_kernel(const _Float16* __restrict__ NEI,
                 const _Float16* __restrict__ Wf,
                 const float*    __restrict__ h0,
                 float*          __restrict__ Hout)
{
    const int lane = threadIdx.x & 31;
    const int wave = threadIdx.x >> 5;
    const int tile = blockIdx.x * 8 + wave;   // 12500 blocks * 8 = 100000 tiles exactly
    const int tn = tile & 31;                  // 32 N tiles (HIDDEN/16)
    const int tm = tile >> 5;                  // 3125 M tiles (N_ATOMS/16)
    const int hsel = lane >> 4;                // 0 or 1 (lane half)
    const int l15  = lane & 15;

    const _Float16* ap = NEI + (size_t)(tm * 16 + l15) * KPAD + hsel * 16;
    const _Float16* bp = Wf  + (size_t)(tn * 16 + l15) * KPAD + hsel * 16;

    v8f c = {};
    #pragma unroll
    for (int k0 = 0; k0 < KPAD; k0 += 32) {
        union { v16h v; v8h h[2]; } au;
        au.h[0] = *(const v8h*)(ap + k0);        // K = k0 + 16*hsel + [0..7]
        au.h[1] = *(const v8h*)(ap + k0 + 8);    // K = k0 + 16*hsel + 8 + [0..7]
        v16h bv = *(const v16h*)(bp + k0);       // K = k0 + 16*hsel + [0..15]
        c = __builtin_amdgcn_wmma_f32_16x16x32_f16(
                false, au.v, false, bv, (short)0, c, false, false);
    }

    const int n = tn * 16 + l15;
    #pragma unroll
    for (int v = 0; v < 8; ++v) {
        int m = tm * 16 + hsel * 8 + v;
        size_t idx = (size_t)m * HIDDEN + n;
        float r = c[v] + h0[idx];
        Hout[idx] = r > 0.f ? r : 0.f;
    }
}

// ---------------------------------------------------------------------------
// out[n][a] = H[a][n]  (LDS-tiled transpose, [HIDDEN x N_ATOMS] output)
// ---------------------------------------------------------------------------
__global__ void __launch_bounds__(256)
transpose_kernel(const float* __restrict__ H, float* __restrict__ out)
{
    __shared__ float t[32][33];
    const int a0 = blockIdx.x * 32;
    const int h0i = blockIdx.y * 32;
    const int tx = threadIdx.x;      // 32
    const int ty = threadIdx.y;      // 8
    for (int i = ty; i < 32; i += 8) {
        int a = a0 + i;
        if (a < N_ATOMS) t[i][tx] = H[(size_t)a * HIDDEN + h0i + tx];
    }
    __syncthreads();
    for (int i = ty; i < 32; i += 8) {
        int h = h0i + i;
        int a = a0 + tx;
        if (a < N_ATOMS) out[(size_t)h * N_ATOMS + a] = t[tx][i];
    }
}

// ---------------------------------------------------------------------------
extern "C" void kernel_launch(void* const* d_in, const int* in_sizes, int n_in,
                              void* d_out, int out_size, void* d_ws, size_t ws_size,
                              hipStream_t stream)
{
    const float* fatoms = (const float*)d_in[0];
    const float* fbonds = (const float*)d_in[1];
    const float* W_nin  = (const float*)d_in[2];
    const float* W_node = (const float*)d_in[3];
    const int*   aout   = (const int*)d_in[4];
    const int*   in_n   = (const int*)d_in[5];
    float* out = (float*)d_out;

    char* ws = (char*)d_ws;
    const size_t H_BYTES   = (size_t)N_ATOMS * HIDDEN * sizeof(float);   // 102,400,000
    const size_t NEI_BYTES = (size_t)N_ATOMS * KPAD * sizeof(_Float16);  //  54,400,000
    float*    h0b = (float*)ws;
    float*    Hb  = (float*)(ws + H_BYTES);
    _Float16* NEI = (_Float16*)(ws + 2 * H_BYTES);
    _Float16* Wf  = (_Float16*)(ws + 2 * H_BYTES + NEI_BYTES);
    // total ~259.8 MB of workspace

    nin_kernel<<<N_ATOMS, 256, 0, stream>>>(fatoms, W_nin, h0b);
    wconv_kernel<<<(HIDDEN * KPAD + 255) / 256, 256, 0, stream>>>(W_node, Wf);

    const float* Hsrc = h0b;
    for (int d = 0; d < DEPTH; ++d) {
        gather_kernel<<<(N_ATOMS + 7) / 8, 256, 0, stream>>>(Hsrc, fbonds, aout, in_n, NEI);
        node_gemm_kernel<<<(N_ATOMS / 16) * (HIDDEN / 16) / 8, 256, 0, stream>>>(NEI, Wf, h0b, Hb);
        Hsrc = Hb;
    }

    transpose_kernel<<<dim3((N_ATOMS + 31) / 32, HIDDEN / 32), dim3(32, 8), 0, stream>>>(Hb, out);
}
